// GraphEncoder_11751030522455
// MI455X (gfx1250) — matrix operand order repaired and checked by reading, not compiled, not run on hardware
//
#include <hip/hip_runtime.h>
#include <hip/hip_bf16.h>
#include <stdint.h>

#define N_NODES 100000
#define N_EDGES 1600000
#define IN_DIM  1433
#define K1PAD   1440      // 45 * 32
#define K1PAIR  (K1PAD/2) // 720 uint pairs per row
#define HID     128
#define EMB     64

typedef __attribute__((ext_vector_type(16))) __bf16 v16bf;
typedef __attribute__((ext_vector_type(8)))  float  v8f;

__device__ __forceinline__ uint32_t pack_bf16(float a, float b) {
  uint32_t ua = __builtin_bit_cast(uint32_t, a);
  uint32_t ub = __builtin_bit_cast(uint32_t, b);
  ua = (ua + 0x7FFFu + ((ua >> 16) & 1u)) >> 16;   // RNE
  ub = (ub + 0x7FFFu + ((ub >> 16) & 1u)) >> 16;
  return (ub << 16) | (ua & 0xFFFFu);
}

union ABReg { v16bf v; uint4 q[2]; };

// ---------------- utility kernels ----------------
__global__ void k_fill(float* __restrict__ p, float val, int n) {
  for (int i = blockIdx.x * blockDim.x + threadIdx.x; i < n; i += gridDim.x * blockDim.x)
    p[i] = val;
}

__global__ void k_zero(float* __restrict__ p, int n) {
  for (int i = blockIdx.x * blockDim.x + threadIdx.x; i < n; i += gridDim.x * blockDim.x)
    p[i] = 0.0f;
}

__global__ void k_deg_edges(const int* __restrict__ dst, float* __restrict__ deg) {
  int e = blockIdx.x * blockDim.x + threadIdx.x;
  if (e < N_EDGES) atomicAdd(&deg[dst[e]], 1.0f);
}

__global__ void k_dinv(float* __restrict__ deg) {
  int i = blockIdx.x * blockDim.x + threadIdx.x;
  if (i < N_NODES) {
    float d = deg[i];
    deg[i] = (d > 0.0f) ? rsqrtf(d) : 0.0f;
  }
}

// W1 [IN_DIM][HID] f32  ->  w1t [HID][K1PAD] bf16 (stored as uint pairs, K-padded)
__global__ void k_convert_w1(const float* __restrict__ W1, uint32_t* __restrict__ w1t) {
  int idx = blockIdx.x * blockDim.x + threadIdx.x;
  if (idx >= HID * K1PAIR) return;
  int n = idx / K1PAIR, kp = idx % K1PAIR;
  int k = kp * 2;
  float f0 = (k     < IN_DIM) ? W1[(long)k       * HID + n] : 0.0f;
  float f1 = (k + 1 < IN_DIM) ? W1[(long)(k + 1) * HID + n] : 0.0f;
  w1t[n * K1PAIR + kp] = pack_bf16(f0, f1);
}

// W2 [HID][EMB] f32 -> w2t [EMB][HID] bf16 (uint pairs)
__global__ void k_convert_w2(const float* __restrict__ W2, uint32_t* __restrict__ w2t) {
  int idx = blockIdx.x * blockDim.x + threadIdx.x;
  if (idx >= EMB * (HID / 2)) return;
  int n = idx / (HID / 2), kp = idx % (HID / 2);
  int k = kp * 2;
  w2t[n * (HID / 2) + kp] = pack_bf16(W2[(long)k * EMB + n], W2[(long)(k + 1) * EMB + n]);
}

// ---------------- GEMM1: h1 = x @ W1  (bf16 WMMA, f32 accum) ----------------
__launch_bounds__(256)
__global__ void k_gemm1(const float* __restrict__ x, const uint32_t* __restrict__ w1t,
                        float* __restrict__ h1) {
  __shared__ uint32_t lA[16 * K1PAIR];           // 46080 B: 16 rows x 1440 bf16
  const int row0 = blockIdx.x * 16;

  // stage + convert A tile (each x element touched exactly once across the grid)
  for (int p = threadIdx.x; p < 16 * K1PAIR; p += 256) {
    int r = p / K1PAIR, kp = p % K1PAIR;
    int k = kp * 2;
    const float* xr = x + (long)(row0 + r) * IN_DIM;
    float f0 = (k     < IN_DIM) ? xr[k]     : 0.0f;
    float f1 = (k + 1 < IN_DIM) ? xr[k + 1] : 0.0f;
    lA[r * K1PAIR + kp] = pack_bf16(f0, f1);
  }
  __syncthreads();

  const int lane = threadIdx.x & 31;
  const int wave = threadIdx.x >> 5;             // 8 waves -> 8 x 16 cols = 128
  const int col0 = wave * 16;
  const int m    = lane & 15;
  const int hiA  = (lane >> 4) * 4;              // ISA A-layout half-wave pair offset
  const int hiB  = (lane >> 4) * 4;              // B half-wave pair offset (same as A)
  const uint32_t* arow = lA  + m * K1PAIR;
  const uint32_t* brow = w1t + (long)(col0 + m) * K1PAIR;

  v8f c = {};
  for (int kk = 0; kk < K1PAD; kk += 32) {
    int kp = kk >> 1;
    ABReg a, b;
    a.q[0] = *(const uint4*)(arow + kp + hiA);       // K 0..7  (or 8..15)
    a.q[1] = *(const uint4*)(arow + kp + 8 + hiA);   // K 16..23 (or 24..31)
    b.q[0] = *(const uint4*)(brow + kp + hiB);       // K 0..7  (or 8..15)
    b.q[1] = *(const uint4*)(brow + kp + 8 + hiB);   // K 16..23 (or 24..31)
    c = __builtin_amdgcn_wmma_f32_16x16x32_bf16(false, a.v, false, b.v,
                                                (short)0, c, false, false);
  }

  const int n    = lane & 15;
  const int mrow = (lane >> 4) * 8;                  // C layout: lanes>=16 hold M=8..15
  float* out = h1 + (long)row0 * HID + col0 + n;
#pragma unroll
  for (int r = 0; r < 8; ++r)
    out[(long)(mrow + r) * HID] = c[r];
}

// ---------------- GEMM2: h2 = relu(agg1+b1) @ W2 ----------------
__launch_bounds__(128)
__global__ void k_gemm2(const float* __restrict__ agg1, const float* __restrict__ b1,
                        const uint32_t* __restrict__ w2t, float* __restrict__ h2) {
  __shared__ uint32_t lA[16 * (HID / 2)];        // 4 KB
  const int row0 = blockIdx.x * 16;

  for (int p = threadIdx.x; p < 16 * (HID / 2); p += 128) {
    int r = p >> 6, kp = p & 63;
    int k = kp * 2;
    const float* ar = agg1 + (long)(row0 + r) * HID;
    float f0 = fmaxf(ar[k]     + b1[k],     0.0f);   // fused bias + ReLU
    float f1 = fmaxf(ar[k + 1] + b1[k + 1], 0.0f);
    lA[p] = pack_bf16(f0, f1);
  }
  __syncthreads();

  const int lane = threadIdx.x & 31;
  const int wave = threadIdx.x >> 5;             // 4 waves -> 64 cols
  const int col0 = wave * 16;
  const int m    = lane & 15;
  const int hiA  = (lane >> 4) * 4;
  const int hiB  = (lane >> 4) * 4;
  const uint32_t* arow = lA  + m * (HID / 2);
  const uint32_t* brow = w2t + (col0 + m) * (HID / 2);

  v8f c = {};
#pragma unroll
  for (int kk = 0; kk < HID; kk += 32) {
    int kp = kk >> 1;
    ABReg a, b;
    a.q[0] = *(const uint4*)(arow + kp + hiA);
    a.q[1] = *(const uint4*)(arow + kp + 8 + hiA);
    b.q[0] = *(const uint4*)(brow + kp + hiB);
    b.q[1] = *(const uint4*)(brow + kp + 8 + hiB);
    c = __builtin_amdgcn_wmma_f32_16x16x32_bf16(false, a.v, false, b.v,
                                                (short)0, c, false, false);
  }

  const int n    = lane & 15;
  const int mrow = (lane >> 4) * 8;
  float* out = h2 + (long)row0 * EMB + col0 + n;
#pragma unroll
  for (int r = 0; r < 8; ++r)
    out[(long)(mrow + r) * EMB] = c[r];
}

// ---------------- edge aggregation: out[dst] += feat[src] * norm ----------------
// one edge per (dim/4)-lane group; self-loops are virtual edges [E, E+N)
__global__ void k_agg(const float* __restrict__ feat, const int* __restrict__ src,
                      const int* __restrict__ dst, const float* __restrict__ dinv,
                      float* __restrict__ out, int dim, int cshift) {
  long t = (long)blockIdx.x * blockDim.x + threadIdx.x;
  long e = t >> cshift;
  int  c = (int)(t & ((1 << cshift) - 1));
  if (e >= (long)N_EDGES + N_NODES) return;
  int s, d;
  if (e < N_EDGES) { s = src[e]; d = dst[e]; }
  else             { s = d = (int)(e - N_EDGES); }
  float nrm = dinv[s] * dinv[d];
  const float4 v = *(const float4*)(feat + (long)s * dim + c * 4);
  float* o = out + (long)d * dim + c * 4;
  atomicAdd(o + 0, v.x * nrm);
  atomicAdd(o + 1, v.y * nrm);
  atomicAdd(o + 2, v.z * nrm);
  atomicAdd(o + 3, v.w * nrm);
}

__global__ void k_bias2(float* __restrict__ out, const float* __restrict__ b2) {
  for (int i = blockIdx.x * blockDim.x + threadIdx.x; i < N_NODES * EMB;
       i += gridDim.x * blockDim.x)
    out[i] += b2[i & (EMB - 1)];
}

static inline size_t alignup(size_t v) { return (v + 255) & ~(size_t)255; }

extern "C" void kernel_launch(void* const* d_in, const int* in_sizes, int n_in,
                              void* d_out, int out_size, void* d_ws, size_t ws_size,
                              hipStream_t stream) {
  const float* x  = (const float*)d_in[0];
  const int*   ei = (const int*)d_in[1];      // [2, E]
  const float* W1 = (const float*)d_in[2];
  const float* b1 = (const float*)d_in[3];
  const float* W2 = (const float*)d_in[4];
  const float* b2 = (const float*)d_in[5];
  float* out = (float*)d_out;

  const int* e_src = ei;
  const int* e_dst = ei + N_EDGES;

  // workspace layout (~129 MB)
  char* ws = (char*)d_ws;
  size_t off = 0;
  float*    dinv = (float*)(ws + off);    off = alignup(off + (size_t)N_NODES * 4);
  uint32_t* w1t  = (uint32_t*)(ws + off); off = alignup(off + (size_t)HID * K1PAIR * 4);
  uint32_t* w2t  = (uint32_t*)(ws + off); off = alignup(off + (size_t)EMB * (HID / 2) * 4);
  float*    h1   = (float*)(ws + off);    off = alignup(off + (size_t)N_NODES * HID * 4);
  float*    agg1 = (float*)(ws + off);    off = alignup(off + (size_t)N_NODES * HID * 4);
  float*    h2   = (float*)(ws + off);    off = alignup(off + (size_t)N_NODES * EMB * 4);
  (void)ws_size; (void)in_sizes; (void)n_in; (void)out_size;

  // degrees -> dinv (deg init 1.0 covers self-loop)
  k_fill<<<391, 256, 0, stream>>>(dinv, 1.0f, N_NODES);
  k_deg_edges<<<(N_EDGES + 255) / 256, 256, 0, stream>>>(e_dst, dinv);
  k_dinv<<<(N_NODES + 255) / 256, 256, 0, stream>>>(dinv);

  // weight conversion (bf16, transposed, K-padded)
  k_convert_w1<<<(HID * K1PAIR + 255) / 256, 256, 0, stream>>>(W1, w1t);
  k_convert_w2<<<(EMB * (HID / 2) + 255) / 256, 256, 0, stream>>>(W2, w2t);

  // layer 1
  k_gemm1<<<N_NODES / 16, 256, 0, stream>>>(x, w1t, h1);
  k_zero<<<2048, 256, 0, stream>>>(agg1, N_NODES * HID);
  {
    long total = ((long)N_EDGES + N_NODES) * (HID / 4);      // 54.4M threads
    k_agg<<<(unsigned)((total + 255) / 256), 256, 0, stream>>>(
        h1, e_src, e_dst, dinv, agg1, HID, 5);
  }

  // layer 2 (bias+ReLU fused into GEMM2 A-staging)
  k_gemm2<<<N_NODES / 16, 128, 0, stream>>>(agg1, b1, w2t, h2);
  k_zero<<<2048, 256, 0, stream>>>(out, N_NODES * EMB);
  {
    long total = ((long)N_EDGES + N_NODES) * (EMB / 4);      // 27.2M threads
    k_agg<<<(unsigned)((total + 255) / 256), 256, 0, stream>>>(
        h2, e_src, e_dst, dinv, out, EMB, 4);
  }
  k_bias2<<<8192, 256, 0, stream>>>(out, b2);
}
